// GRUCell_12317966205535
// MI455X (gfx1250) — compile-verified, matrix operand order checked by reading, not run on hardware
//
#include <hip/hip_runtime.h>
#include <hip/hip_bf16.h>

typedef __attribute__((ext_vector_type(16))) __bf16 v16bf;
typedef __attribute__((ext_vector_type(8)))  float  v8f;

union FragAB { v16bf v; uint4 q[2]; };

// float -> bf16 bits, round-to-nearest-even (pure integer, no __bf16 scalar ops)
__device__ __forceinline__ unsigned short f2bf(float f) {
    unsigned u = __float_as_uint(f);
    u += 0x7FFFu + ((u >> 16) & 1u);
    return (unsigned short)(u >> 16);
}

__device__ __forceinline__ float sigmoidf(float v) {
    return 1.0f / (1.0f + __expf(-v));
}

// ---------------- zero workspace ----------------
__global__ void __launch_bounds__(256) k_zero(float* p, long long n) {
    long long i = (long long)blockIdx.x * 256 + threadIdx.x;
    if (i < n) p[i] = 0.0f;
}

// ---------------- degree ----------------
__global__ void __launch_bounds__(256) k_deg(const int* dst, float* deg, long long E) {
    long long e = (long long)blockIdx.x * 256 + threadIdx.x;
    if (e < E) atomicAdd(&deg[dst[e]], 1.0f);
}

// ---------------- norm = rsqrt(max(deg,1)) in place ----------------
__global__ void __launch_bounds__(256) k_norm(float* deg, int N) {
    int i = blockIdx.x * 256 + threadIdx.x;
    if (i < N) deg[i] = rsqrtf(fmaxf(deg[i], 1.0f));
}

// ---------------- edge scatter: agg[dst] += feat[src]*norm[src] ----------------
// one wave32 per edge; each lane owns 4 floats of x and 4 of hx
__global__ void __launch_bounds__(256) k_scatter(const int* src, const int* dst,
                                                 const float* norm,
                                                 const float* x, const float* hx,
                                                 float* aggx, float* aggh, long long E) {
    long long t = (long long)blockIdx.x * 256 + threadIdx.x;
    long long e = t >> 5;
    int lane = (int)(t & 31);
    if (e >= E) return;
    int s = src[e], d = dst[e];
    float ns = norm[s];
    float4 xv = ((const float4*)(x  + (long long)s * 128))[lane];
    float4 hv = ((const float4*)(hx + (long long)s * 128))[lane];
    float* ax = aggx + (long long)d * 128 + lane * 4;
    float* ah = aggh + (long long)d * 128 + lane * 4;
    atomicAdd(ax + 0, xv.x * ns); atomicAdd(ax + 1, xv.y * ns);
    atomicAdd(ax + 2, xv.z * ns); atomicAdd(ax + 3, xv.w * ns);
    atomicAdd(ah + 0, hv.x * ns); atomicAdd(ah + 1, hv.y * ns);
    atomicAdd(ah + 2, hv.z * ns); atomicAdd(ah + 3, hv.w * ns);
}

// ---------------- pack stacked weights (bf16, K-contiguous per output col) ----------
// WT[col][k], col in [0,512): gate=col>>7, h=col&127
// k in [0,512): seg = k>>7 (0:x T0, 1:px T1, 2:hx T0, 3:ph T1), f = k&127
__global__ void __launch_bounds__(256) k_prepw(const float* Wrx, const float* Wrh,
                                               const float* Wux, const float* Wuh,
                                               const float* Wcx, const float* Wch,
                                               const float* brx, const float* brh,
                                               const float* bux, const float* buh,
                                               const float* bcx, const float* bch,
                                               unsigned short* WT, float* bias) {
    int i = blockIdx.x * 256 + threadIdx.x;     // 0 .. 512*512-1
    int col = i >> 9;
    int k = i & 511;
    int g = col >> 7, h = col & 127;
    int seg = k >> 7, f = k & 127;
    int t = seg & 1;          // Chebyshev order (T0 / T1)
    int hs = seg >> 1;        // 0 = x-side, 1 = h-side
    const float* W = nullptr;
    if (g == 0)      W = hs ? Wrh : Wrx;
    else if (g == 1) W = hs ? Wuh : Wux;
    else if (g == 2) W = hs ? nullptr : Wcx;    // cx gets only x-side
    else             W = hs ? Wch : nullptr;    // ch gets only h-side
    float v = W ? W[(t * 128 + f) * 128 + h] : 0.0f;
    WT[col * 512 + k] = f2bf(v);
    if (i < 512) {
        int gg = i >> 7, hh = i & 127;
        float b;
        if (gg == 0)      b = brx[hh] + brh[hh];
        else if (gg == 1) b = bux[hh] + buh[hh];
        else if (gg == 2) b = bcx[hh];
        else              b = bch[hh];
        bias[i] = b;
    }
}

// ---------------- fused WMMA GEMM + GRU gates ----------------
// Block: 256 threads (8 waves). One block = 32 node rows x 512 output cols.
// Each wave owns TWO 16-row tiles sharing every B fragment (2 WMMAs / B load pair),
// with B fragments software-pipelined one step ahead so L2 latency hides under WMMA.
// LDS: A-panel (32x520 bf16) overlaid with 16x512 f32 D-stage (A dead after K loop).
__global__ void __launch_bounds__(256) k_gemm_gates(const float* x, const float* hx,
                                                    const float* aggx, const float* aggh,
                                                    const float* norm,
                                                    const unsigned short* WT,
                                                    const float* bias,
                                                    float* hy, int N) {
    __shared__ __align__(16) union SM {
        unsigned short A[32][520];   // +8 pad: rows shift 16B -> conflict-free ds_load_b128
        float D[16][512];            // reused after K loop, one 16-row tile at a time
    } sm;

    const int base = blockIdx.x * 32;
    const int tid = threadIdx.x;
    const int lane = tid & 31;
    const int wave = tid >> 5;
    const int l15 = lane & 15;
    const int half = lane >> 4;

    // B fragment loader: WT is K-contiguous per column -> two 16B loads (L2-resident)
    const unsigned short* wtLane = WT + (wave * 64 + l15) * 512 + half * 8;
    auto loadB = [&](int ks, int j, FragAB& b) {
        const unsigned short* p = wtLane + j * 16 * 512 + ks;
        b.q[0] = *(const uint4*)p;
        b.q[1] = *(const uint4*)(p + 16);
    };

    // preload first B fragment: latency hides under the whole A-panel fill
    FragAB bcur;
    loadB(0, 0, bcur);

    // cooperative A-panel fill (32 rows x 512 K), ChebConv T1 scaling fused in
    for (int i = tid; i < 32 * 512; i += 256) {
        int r = i >> 9;
        int k = i & 511;
        int node = base + r;
        float v = 0.0f;
        if (node < N) {
            int seg = k >> 7, f = k & 127;
            long long o = (long long)node * 128 + f;
            if (seg == 0)      v = x[o];
            else if (seg == 1) v = -aggx[o] * norm[node];
            else if (seg == 2) v = hx[o];
            else               v = -aggh[o] * norm[node];
        }
        sm.A[r][k] = f2bf(v);
    }
    __syncthreads();

    v8f acc[2][4] = {};   // [row tile][column tile], 16x16 each

    for (int ks = 0; ks < 512; ks += 32) {
        // A fragments: 16-bit A 16x32 layout (lanes 0-15: K 0..7 & 16..23; lanes 16-31: K 8..15 & 24..31)
        FragAB a0, a1;
        a0.q[0] = *(const uint4*)&sm.A[l15][ks + half * 8];
        a0.q[1] = *(const uint4*)&sm.A[l15][ks + 16 + half * 8];
        a1.q[0] = *(const uint4*)&sm.A[16 + l15][ks + half * 8];
        a1.q[1] = *(const uint4*)&sm.A[16 + l15][ks + 16 + half * 8];
#pragma unroll
        for (int j = 0; j < 4; ++j) {
            // issue next B load BEFORE consuming the current fragment (software pipeline);
            // final step wraps in-bounds to stay branch-free
            FragAB bnext;
            int nj  = (j == 3) ? 0 : j + 1;
            int nks = (j == 3) ? ((ks + 32) & 511) : ks;
            loadB(nks, nj, bnext);
            acc[0][j] = __builtin_amdgcn_wmma_f32_16x16x32_bf16(
                false, a0.v, false, bcur.v, (short)0, acc[0][j], false, false);
            acc[1][j] = __builtin_amdgcn_wmma_f32_16x16x32_bf16(
                false, a1.v, false, bcur.v, (short)0, acc[1][j], false, false);
            bcur = bnext;
        }
    }
    __syncthreads();   // all waves done reading sm.A; safe to overlay sm.D

    // gate epilogue, one 16-row tile at a time
#pragma unroll
    for (int rt = 0; rt < 2; ++rt) {
        // D tile layout: VGPR r, lanes 0-15 -> M=r, lanes 16-31 -> M=r+8
#pragma unroll
        for (int j = 0; j < 4; ++j) {
            int col = wave * 64 + j * 16 + l15;
#pragma unroll
            for (int r = 0; r < 8; ++r) {
                sm.D[r + half * 8][col] = acc[rt][j][r];
            }
        }
        __syncthreads();

        for (int i = tid; i < 16 * 128; i += 256) {
            int rr = i >> 7, h = i & 127;
            int node = base + rt * 16 + rr;
            if (node < N) {
                float gr  = sm.D[rr][h]       + bias[h];
                float gu  = sm.D[rr][128 + h] + bias[128 + h];
                float gcx = sm.D[rr][256 + h] + bias[256 + h];
                float gch = sm.D[rr][384 + h] + bias[384 + h];
                float r_ = sigmoidf(gr);
                float u_ = sigmoidf(gu);
                float c_ = sigmoidf(gcx + gch * r_);  // reset multiplies h-side cheb (incl. b_ch)
                long long o = (long long)node * 128 + h;
                float h0 = hx[o];
                hy[o] = u_ * h0 + (1.0f - u_) * c_;
            }
        }
        __syncthreads();
    }
}

extern "C" void kernel_launch(void* const* d_in, const int* in_sizes, int n_in,
                              void* d_out, int out_size, void* d_ws, size_t ws_size,
                              hipStream_t stream) {
    const int*   ei  = (const int*)d_in[0];
    const float* x   = (const float*)d_in[1];
    const float* hx  = (const float*)d_in[2];
    const float* Wrx = (const float*)d_in[3];  const float* brx = (const float*)d_in[4];
    const float* Wrh = (const float*)d_in[5];  const float* brh = (const float*)d_in[6];
    const float* Wux = (const float*)d_in[7];  const float* bux = (const float*)d_in[8];
    const float* Wuh = (const float*)d_in[9];  const float* buh = (const float*)d_in[10];
    const float* Wcx = (const float*)d_in[11]; const float* bcx = (const float*)d_in[12];
    const float* Wch = (const float*)d_in[13]; const float* bch = (const float*)d_in[14];

    const long long E = (long long)in_sizes[0] / 2;
    const int N = in_sizes[1] / 128;

    // workspace layout (256B aligned)
    char* ws = (char*)d_ws;
    size_t degBytes = ((size_t)N * 4 + 255) & ~(size_t)255;
    size_t aggBytes = (size_t)N * 128 * 4;
    float* deg  = (float*)ws;                                  // reused as norm
    float* aggx = (float*)(ws + degBytes);
    float* aggh = (float*)(ws + degBytes + aggBytes);
    unsigned short* WT = (unsigned short*)(ws + degBytes + 2 * aggBytes);
    float* bias = (float*)(ws + degBytes + 2 * aggBytes + (size_t)512 * 512 * 2);

    const int* src = ei;
    const int* dst = ei + E;

    long long zc = (long long)((degBytes + 2 * aggBytes) / 4);
    k_zero<<<(unsigned)((zc + 255) / 256), 256, 0, stream>>>((float*)ws, zc);
    k_deg<<<(unsigned)((E + 255) / 256), 256, 0, stream>>>(dst, deg, E);
    k_norm<<<(N + 255) / 256, 256, 0, stream>>>(deg, N);
    k_scatter<<<(unsigned)((E * 32 + 255) / 256), 256, 0, stream>>>(
        src, dst, deg, x, hx, aggx, aggh, E);
    k_prepw<<<(512 * 512) / 256, 256, 0, stream>>>(
        Wrx, Wrh, Wux, Wuh, Wcx, Wch, brx, brh, bux, buh, bcx, bch, WT, bias);
    k_gemm_gates<<<(N + 31) / 32, 256, 0, stream>>>(
        x, hx, aggx, aggh, deg, WT, bias, (float*)d_out, N);
}
